// ForecastModel_5394478924222
// MI455X (gfx1250) — matrix operand order repaired and checked by reading, not compiled, not run on hardware
//
#include <hip/hip_runtime.h>
#include <hip/hip_bf16.h>
#include <cstdint>
#include <cstddef>

// ---------------------------------------------------------------------------
// Model dims
// ---------------------------------------------------------------------------
#define H1v   512
#define G4v   2048          // 4*H1
#define H2v   256
#define H3v   128
#define Bv    1024
#define Sv    128
#define CINv  6
#define NBITS 7
#define KEXT0 64            // 42 bits padded to 64

typedef __attribute__((ext_vector_type(16))) __bf16 v16bf;
typedef __attribute__((ext_vector_type(8)))  __bf16 v8bf;
typedef __attribute__((ext_vector_type(8)))  float  v8f;

static __device__ __forceinline__ float sigf(float x) { return 1.0f / (1.0f + __expf(-x)); }

static __device__ __forceinline__ v16bf make_frag(const __bf16* p0, const __bf16* p1) {
    v8bf x0 = *(const v8bf*)p0;
    v8bf x1 = *(const v8bf*)p1;
    v16bf r;
#pragma unroll
    for (int i = 0; i < 8; ++i) { r[i] = x0[i]; r[i + 8] = x1[i]; }
    return r;
}

// CDNA5 async global->LDS copy (16B per lane), tracked by ASYNCcnt.
static __device__ __forceinline__ void async_ld16(uint32_t lds_byte_addr, const void* gptr) {
    asm volatile("global_load_async_to_lds_b128 %0, %1, off"
                 :
                 : "v"(lds_byte_addr), "v"((uint64_t)(uintptr_t)gptr)
                 : "memory");
}
static __device__ __forceinline__ void wait_async0() {
    asm volatile("s_wait_asynccnt 0x0" ::: "memory");
}

// ---------------------------------------------------------------------------
// Precompute kernels (run every launch; deterministic)
// ---------------------------------------------------------------------------
__global__ void k_f32_to_bf16(const float* __restrict__ src, __bf16* __restrict__ dst, int n) {
    for (int i = blockIdx.x * blockDim.x + threadIdx.x; i < n; i += gridDim.x * blockDim.x)
        dst[i] = (__bf16)src[i];
}

__global__ void k_zero_u32(uint32_t* __restrict__ p, int n) {
    for (int i = blockIdx.x * blockDim.x + threadIdx.x; i < n; i += gridDim.x * blockDim.x)
        p[i] = 0u;
}

// Weff[n,k] = sum_j Wih0[n,j] * W1[j,k]   (k<42, padded to 64 with zeros), bf16
__global__ void k_weff(const float* __restrict__ Wih0, const float* __restrict__ W1,
                       __bf16* __restrict__ Weff) {
    int idx = blockIdx.x * blockDim.x + threadIdx.x;
    if (idx >= G4v * KEXT0) return;
    int n = idx >> 6, k = idx & 63;
    float s = 0.0f;
    if (k < CINv * NBITS) {
        const float* wr = Wih0 + (size_t)n * H1v;
        for (int j = 0; j < H1v; ++j) s += wr[j] * W1[j * (CINv * NBITS) + k];
    }
    Weff[idx] = (__bf16)s;
}

// beff[n] = Wih0[n,:]·b1 + bih0[n] + bhh0[n]
__global__ void k_beff(const float* __restrict__ Wih0, const float* __restrict__ b1,
                       const float* __restrict__ bih0, const float* __restrict__ bhh0,
                       float* __restrict__ beff) {
    int n = blockIdx.x * blockDim.x + threadIdx.x;
    if (n >= G4v) return;
    float s = bih0[n] + bhh0[n];
    const float* wr = Wih0 + (size_t)n * H1v;
    for (int j = 0; j < H1v; ++j) s += wr[j] * b1[j];
    beff[n] = s;
}

__global__ void k_bias_add(const float* __restrict__ a, const float* __restrict__ b,
                           float* __restrict__ o, int n) {
    int i = blockIdx.x * blockDim.x + threadIdx.x;
    if (i < n) o[i] = a[i] + b[i];
}

// ---------------------------------------------------------------------------
// LSTM step kernel: one timestep of one layer.
//   gates[M=1024, 4*512] = hprev(bf16) @ Whh(bf16)^T  +  Aext @ Bext^T  + bias
//   then fused i,f,g,o activation + cell/hidden update.
// Block tile: 128 rows x 32 hidden units (x 4 gates). 8 waves, each wave does
// a 32x16 subtile per gate => 8 accumulators x v8f, 8 WMMAs per K-tile of 32.
// Tiles stream through double-buffered LDS via GLOBAL_LOAD_ASYNC_TO_LDS_B128
// (ASYNCcnt), one barrier per K-tile.
// L0: last 2 tiles' A operand is bit-expanded x[:,t,:] generated into LDS.
// L1: Aext = h0 of this step (bf16, K=512, Wih1).
// ---------------------------------------------------------------------------
#define LDT   40             // LDS row stride in bf16 (80B: 16B-aligned chunks)
#define ABUFE (128 * LDT)    // elements per A buffer
#define BBUFE (4 * 32 * LDT) // elements per B buffer

template <bool L0>
__global__ __launch_bounds__(256) void lstm_step_kernel(
    const __bf16* __restrict__ hprev, const __bf16* __restrict__ Whh,
    const __bf16* __restrict__ Aext,  const __bf16* __restrict__ Bext,
    const int*    __restrict__ xin,   int t,
    const float*  __restrict__ bias,  float* __restrict__ c,
    __bf16* __restrict__ hout,        float* __restrict__ hf32)
{
    __shared__ __bf16 ldsA[2 * ABUFE];
    __shared__ __bf16 ldsB[2 * BBUFE];

    const int tid = threadIdx.x;
    const int m0 = blockIdx.x * 128;
    const int n0 = blockIdx.y * 32;
    const int w = tid >> 5, lane = tid & 31;
    const int wm = w & 3, wn = w >> 2;            // 4 waves in M, 2 in N
    const int r16 = lane & 15, hi = lane >> 4;

    v8f acc[4][2];
#pragma unroll
    for (int g = 0; g < 4; ++g)
#pragma unroll
        for (int mt = 0; mt < 2; ++mt) acc[g][mt] = {};

    const int NT = L0 ? 18 : 32;   // 16 recurrent K-tiles + input K-tiles

    // Issue all loads/stores that populate K-tile i into buffer bsel.
    auto issue = [&](int i, int bsel) {
        __bf16* dA = ldsA + bsel * ABUFE;
        __bf16* dB = ldsB + bsel * BBUFE;
        const __bf16* sA = hprev;
        const __bf16* sB = Whh;
        int strideB = H1v, kb;
        bool gen = false;
        if (i < 16) {
            kb = i * 32;
        } else {
            kb = (i - 16) * 32;
            sB = Bext;
            if constexpr (L0) { gen = true; strideB = KEXT0; }
            else              { sA = Aext; }
        }
        // B tile: 4 gates x 32 rows x 32 bf16 = 512 x 16B chunks, 2 per thread
#pragma unroll
        for (int q = 0; q < 2; ++q) {
            int chunk = tid + q * 256;
            int rr = chunk >> 2, cc = chunk & 3;
            int g = rr >> 5, n = rr & 31;
            async_ld16((uint32_t)(uintptr_t)(dB + (g * 32 + n) * LDT + cc * 8),
                       sB + (size_t)(g * H1v + n0 + n) * strideB + kb + cc * 8);
        }
        if (!gen) {
            // A tile: 128 rows x 32 bf16 = 512 x 16B chunks, 2 per thread
#pragma unroll
            for (int q = 0; q < 2; ++q) {
                int chunk = tid + q * 256;
                int r = chunk >> 2, cc = chunk & 3;
                async_ld16((uint32_t)(uintptr_t)(dA + r * LDT + cc * 8),
                           sA + (size_t)(m0 + r) * H1v + kb + cc * 8);
            }
        } else {
            // L0 input tile: bit-expand x[:, t, :] straight into LDS (DScnt,
            // fenced by the per-iteration __syncthreads()).
            int r = tid >> 1, half = tid & 1;
            int xbase = (int)(((size_t)(m0 + r) * Sv + t) * CINv);
#pragma unroll
            for (int j = 0; j < 16; ++j) {
                int jj = half * 16 + j;
                int k = kb + jj;
                float v = 0.0f;
                if (k < CINv * NBITS) {
                    int ci = k / NBITS, bi = k % NBITS;
                    v = (float)((xin[xbase + ci] >> bi) & 1);
                }
                dA[r * LDT + jj] = (__bf16)v;
            }
        }
    };

    auto mma_all = [&](int bsel) {
        const __bf16* A = ldsA + bsel * ABUFE;
        const __bf16* Bt = ldsB + bsel * BBUFE;
        v16bf a[2];
#pragma unroll
        for (int mt = 0; mt < 2; ++mt) {
            int ro = (wm * 32 + mt * 16 + r16) * LDT;
            a[mt] = make_frag(&A[ro + 16 * hi], &A[ro + 16 * hi + 8]);
        }
#pragma unroll
        for (int g = 0; g < 4; ++g) {
            int rb = (g * 32 + wn * 16 + r16) * LDT + 16 * hi;
            v16bf b = make_frag(&Bt[rb], &Bt[rb + 8]);
#pragma unroll
            for (int mt = 0; mt < 2; ++mt)
                acc[g][mt] = __builtin_amdgcn_wmma_f32_16x16x32_bf16(
                    false, a[mt], false, b, (short)0, acc[g][mt], false, false);
        }
    };

    // Software pipeline: tile i computes while tile i+1 streams in.
    int buf = 0;
    issue(0, 0);
    for (int i = 0; i < NT; ++i) {
        wait_async0();            // this wave's tile-i chunks have landed
        __syncthreads();          // everyone's landed; other buffer now free
        if (i + 1 < NT) issue(i + 1, buf ^ 1);
        mma_all(buf);
        buf ^= 1;
    }

    // ---- fused gate epilogue: c' = sig(f)*c + sig(i)*tanh(g); h = sig(o)*tanh(c') ----
#pragma unroll
    for (int mt = 0; mt < 2; ++mt)
#pragma unroll
        for (int v = 0; v < 8; ++v) {
            int m = m0 + wm * 32 + mt * 16 + v + 8 * hi;
            int n = n0 + wn * 16 + r16;
            size_t idx = (size_t)m * H1v + n;
            float iv = acc[0][mt][v] + bias[n];
            float fv = acc[1][mt][v] + bias[H1v + n];
            float gv = acc[2][mt][v] + bias[2 * H1v + n];
            float ov = acc[3][mt][v] + bias[3 * H1v + n];
            float cn = sigf(fv) * c[idx] + sigf(iv) * tanhf(gv);
            float hn = sigf(ov) * tanhf(cn);
            c[idx] = cn;
            hout[idx] = (__bf16)hn;
            if (hf32) hf32[idx] = hn;
        }
}

// ---------------------------------------------------------------------------
// Head-side kernels (tiny fraction of FLOPs; simple VALU kernels)
// ---------------------------------------------------------------------------
__global__ void k_gemm_f32(const float* __restrict__ A, const float* __restrict__ W,
                           const float* __restrict__ bias, float* __restrict__ out,
                           int M, int N, int K) {
    int n = blockIdx.x * 16 + threadIdx.x;
    int m = blockIdx.y * 16 + threadIdx.y;
    if (m >= M || n >= N) return;
    float s = bias ? bias[n] : 0.0f;
    const float* a = A + (size_t)m * K;
    const float* wr = W + (size_t)n * K;
    for (int k = 0; k < K; ++k) s += a[k] * wr[k];
    out[(size_t)m * N + n] = s;
}

// h2 = tanh(lin2) * sigmoid(am2pre * topk_mask(lin2, floor(kk*256)))
__global__ void k_h2(const float* __restrict__ lin2, const float* __restrict__ am2p,
                     const float* __restrict__ kk, float* __restrict__ h2) {
    int m = blockIdx.x, j = threadIdx.x;            // 256 threads
    __shared__ float v[H2v];
    float x = lin2[(size_t)m * H2v + j];
    v[j] = x;
    __syncthreads();
    int kq = (int)floorf(kk[0] * (float)H2v);
    int rank = 0;
    for (int i = 0; i < H2v; ++i) {
        float vi = v[i];
        rank += (vi > x) || (vi == x && i < j);     // argsort tie-break by index
    }
    float mask = (rank < kq) ? 1.0f : 0.0f;
    float a2 = sigf(am2p[(size_t)m * H2v + j] * mask);
    h2[(size_t)m * H2v + j] = tanhf(x) * a2;
}

// one output head: am/z GEMMs (K=256), topk mask, astro, dropout, tanh, dot-reduce
__global__ void k_head(const float* __restrict__ h2,
                       const float* __restrict__ Wam, const float* __restrict__ bam,
                       const float* __restrict__ Wlin, const float* __restrict__ blin,
                       const float* __restrict__ Whead, const float* __restrict__ bhead,
                       const float* __restrict__ kp, const float* __restrict__ amp,
                       const float* __restrict__ ampn, int head, float* __restrict__ out) {
    int m = blockIdx.x, j = threadIdx.x;            // 128 threads
    __shared__ float hrow[H2v];
    __shared__ float amv[H3v];
    __shared__ float red[H3v];
    hrow[j] = h2[(size_t)m * H2v + j];
    hrow[j + H3v] = h2[(size_t)m * H2v + H3v + j];
    __syncthreads();
    float am = bam[j], z = blin[j];
    const float* wa = Wam + (size_t)j * H2v;
    const float* wl = Wlin + (size_t)j * H2v;
    for (int k = 0; k < H2v; ++k) { float hv = hrow[k]; am += hv * wa[k]; z += hv * wl[k]; }
    amv[j] = am;
    __syncthreads();
    int kq = (int)floorf(kp[0] * (float)H3v);
    int rank = 0;
    for (int i = 0; i < H3v; ++i) {
        float vi = amv[i];
        rank += (vi > am) || (vi == am && i < j);
    }
    float mask = (rank < kq) ? 1.0f : 0.0f;
    float astro = sigf(am * mask) * amp[0] * ampn[0] * 100.0f;
    // deterministic dropout (p=0.1), splitmix64 hash of (head, element)
    uint64_t s = ((uint64_t)(head + 1) << 32) ^ (uint64_t)((size_t)m * H3v + j);
    s += 0x9E3779B97F4A7C15ull;
    s ^= s >> 30; s *= 0xBF58476D1CE4E5B9ull;
    s ^= s >> 27; s *= 0x94D049BB133111EBull;
    s ^= s >> 31;
    float u = (float)(s >> 40) * (1.0f / 16777216.0f);
    z = (u < 0.9f) ? z * (1.0f / 0.9f) : 0.0f;
    red[j] = tanhf(z) * astro * Whead[j];
    __syncthreads();
    for (int st = 64; st > 0; st >>= 1) {
        if (j < st) red[j] += red[j + st];
        __syncthreads();
    }
    if (j == 0) out[m] = (red[0] + bhead[0]) * fmaxf(ampn[0] * 10.0f, 0.0f);
}

// ---------------------------------------------------------------------------
// Host launcher
// ---------------------------------------------------------------------------
extern "C" void kernel_launch(void* const* d_in, const int* in_sizes, int n_in,
                              void* d_out, int out_size, void* d_ws, size_t ws_size,
                              hipStream_t stream) {
    // input index map (setup_inputs dict order)
    const int*   x     = (const int*)  d_in[0];
    const float* W1    = (const float*)d_in[1];
    const float* b1    = (const float*)d_in[2];
    const float* Wih0  = (const float*)d_in[3];
    const float* Whh0  = (const float*)d_in[4];
    const float* bih0  = (const float*)d_in[5];
    const float* bhh0  = (const float*)d_in[6];
    const float* Wih1  = (const float*)d_in[7];
    const float* Whh1  = (const float*)d_in[8];
    const float* bih1  = (const float*)d_in[9];
    const float* bhh1  = (const float*)d_in[10];
    const float* W2    = (const float*)d_in[11];
    const float* b2    = (const float*)d_in[12];
    const float* Wam2  = (const float*)d_in[13];
    const float* bam2  = (const float*)d_in[14];
    // per-head blocks: 15 + 6*i .. 20 + 6*i  = Wam,bam,Wlin,blin,Whead,bhead
    const float* kk    = (const float*)d_in[51];
    // kp A..F: 52..57 ; amp: 58 ; ampA..F: 59..64
    const float* amp   = (const float*)d_in[58];
    float* out = (float*)d_out;

    // workspace carve-up (256B aligned)
    char* w = (char*)d_ws;
    size_t off = 0;
    auto carve = [&](size_t bytes) {
        void* p = w + off;
        off = (off + bytes + 255) & ~(size_t)255;
        return p;
    };
    const size_t WHH_B = (size_t)G4v * H1v * sizeof(__bf16);     // 2 MB
    __bf16* Whh0b = (__bf16*)carve(WHH_B);
    __bf16* Whh1b = (__bf16*)carve(WHH_B);
    __bf16* Wih1b = (__bf16*)carve(WHH_B);
    __bf16* Weffb = (__bf16*)carve((size_t)G4v * KEXT0 * sizeof(__bf16));
    float*  beff  = (float*)carve(G4v * sizeof(float));
    float*  bias1 = (float*)carve(G4v * sizeof(float));
    // zeroed region (contiguous): c0, c1, h0 ping/pong, h1 ping/pong
    char* zero_start = w + off;
    float*  c0  = (float*)carve((size_t)Bv * H1v * sizeof(float));
    float*  c1  = (float*)carve((size_t)Bv * H1v * sizeof(float));
    __bf16* h0a = (__bf16*)carve((size_t)Bv * H1v * sizeof(__bf16));
    __bf16* h0b = (__bf16*)carve((size_t)Bv * H1v * sizeof(__bf16));
    __bf16* h1a = (__bf16*)carve((size_t)Bv * H1v * sizeof(__bf16));
    __bf16* h1b = (__bf16*)carve((size_t)Bv * H1v * sizeof(__bf16));
    size_t zero_bytes = (size_t)((w + off) - zero_start);
    float* hf1  = (float*)carve((size_t)Bv * H1v * sizeof(float));
    float* lin2 = (float*)carve((size_t)Bv * H2v * sizeof(float));
    float* am2p = (float*)carve((size_t)Bv * H2v * sizeof(float));
    float* h2   = (float*)carve((size_t)Bv * H2v * sizeof(float));

    // --- precompute: bf16 weights, folded input weight, biases, zero state ---
    k_f32_to_bf16<<<1024, 256, 0, stream>>>(Whh0, Whh0b, G4v * H1v);
    k_f32_to_bf16<<<1024, 256, 0, stream>>>(Whh1, Whh1b, G4v * H1v);
    k_f32_to_bf16<<<1024, 256, 0, stream>>>(Wih1, Wih1b, G4v * H1v);
    k_weff<<<(G4v * KEXT0 + 255) / 256, 256, 0, stream>>>(Wih0, W1, Weffb);
    k_beff<<<(G4v + 255) / 256, 256, 0, stream>>>(Wih0, b1, bih0, bhh0, beff);
    k_bias_add<<<(G4v + 255) / 256, 256, 0, stream>>>(bih1, bhh1, bias1, G4v);
    k_zero_u32<<<4096, 256, 0, stream>>>((uint32_t*)zero_start, (int)(zero_bytes / 4));

    // --- recurrence: 128 steps, 2 layers, kernel-per-step (graph amortizes) ---
    dim3 grid(Bv / 128, H1v / 32);   // 8 x 16 blocks, 256 threads
    for (int t = 0; t < Sv; ++t) {
        const __bf16* h0r = (t & 1) ? h0b : h0a;
        __bf16*       h0w = (t & 1) ? h0a : h0b;
        lstm_step_kernel<true><<<grid, 256, 0, stream>>>(
            h0r, Whh0b, nullptr, Weffb, x, t, beff, c0, h0w, nullptr);
        const __bf16* h1r = (t & 1) ? h1b : h1a;
        __bf16*       h1w = (t & 1) ? h1a : h1b;
        lstm_step_kernel<false><<<grid, 256, 0, stream>>>(
            h1r, Whh1b, h0w, Wih1b, nullptr, t, bias1, c1, h1w, hf1);
    }

    // --- heads ---
    k_gemm_f32<<<dim3(H2v / 16, Bv / 16), dim3(16, 16), 0, stream>>>(hf1, W2, b2, lin2, Bv, H2v, H1v);
    k_gemm_f32<<<dim3(H2v / 16, Bv / 16), dim3(16, 16), 0, stream>>>(hf1, Wam2, bam2, am2p, Bv, H2v, H1v);
    k_h2<<<Bv, H2v, 0, stream>>>(lin2, am2p, kk, h2);
    for (int i = 0; i < 6; ++i) {
        const float* Wam   = (const float*)d_in[15 + 6 * i];
        const float* bam   = (const float*)d_in[16 + 6 * i];
        const float* Wlin  = (const float*)d_in[17 + 6 * i];
        const float* blin  = (const float*)d_in[18 + 6 * i];
        const float* Whead = (const float*)d_in[19 + 6 * i];
        const float* bhead = (const float*)d_in[20 + 6 * i];
        const float* kp    = (const float*)d_in[52 + i];
        const float* ampn  = (const float*)d_in[59 + i];
        k_head<<<Bv, H3v, 0, stream>>>(h2, Wam, bam, Wlin, blin, Whead, bhead,
                                       kp, amp, ampn, i, out + (size_t)i * Bv);
    }
    (void)in_sizes; (void)n_in; (void)out_size; (void)ws_size;
}
